// RelativeAbsolutePositionEncodingModel_83519934038298
// MI455X (gfx1250) — matrix operand order repaired and checked by reading, not compiled, not run on hardware
//
#include <hip/hip_runtime.h>
#include <hip/hip_bf16.h>

// ---------------------------------------------------------------------------
// RelativeAbsolutePositionEncodingModel, fused for MI455X (gfx1250, wave32).
// B=131072, L=9, D=22, H=2, FF=128, NB=19.
// One workgroup = 16 batch items = 144 rows = 9 waves x one 16-row M-tile.
// All GEMMs via v_wmma_f32_16x16x32_f16; softmax/LN/head on VALU.
// ---------------------------------------------------------------------------

typedef __attribute__((ext_vector_type(16))) _Float16 v16h;
typedef __attribute__((ext_vector_type(8)))  float    v8f;

#define NFRAG 29
#define FRAG_HALFS (NFRAG * 512)               // 29 frags x 32 lanes x 16 halves
#define WS_PE_OFF  (FRAG_HALFS * 2)            // bytes
#define WS_PB_OFF  (WS_PE_OFF + 9 * 22 * 4)    // bytes

// LDS pool (floats). Aliased across pipeline stages (barriers between).
#define SM_XS   0          // [144][24]  x (then x1, then x2, in place)
#define SM_QS   3456       // [144][48]  q            | later: ff [144][132]
#define SM_KS   10368      // [144][48]  k            | later: y1 [144][24], pr[144]
#define SM_VS   17280      // [144][48]  v
#define SM_HS   24192      // [144][48]  heads        | later: y2 [144][24]
#define SM_FLOATS 31104    // 124416 bytes

__device__ inline v8f wmma16(v16h a, v16h b, v8f c) {
  return __builtin_amdgcn_wmma_f32_16x16x32_f16(false, a, false, b, (short)0, c,
                                                false, false);
}

// Build a 16x32 f16 A-fragment from fp32 LDS rows per CDNA5 wave32 layout:
// lane g=lane>>4, m=lane&15 (row = rowbase+m); half h -> K = 16*(h>>3)+(h&7)+8*g.
__device__ inline v16h load_a_frag(const float* __restrict__ src, int ld,
                                   int rowbase, int lane, int kbase, int kvalid) {
  const int g = lane >> 4;
  const float* row = src + (rowbase + (lane & 15)) * ld;
  v16h a;
#pragma unroll
  for (int h = 0; h < 16; ++h) {
    const int K = ((h >> 3) << 4) + (h & 7) + (g << 3);
    const int k = kbase + K;
    a[h] = (k < kvalid) ? (_Float16)row[k] : (_Float16)0.0f;
  }
  return a;
}

// ---------------------------------------------------------------------------
// Setup kernel: sinusoidal PE, relative-position attention bias, and weight
// matrices pre-packed into WMMA B-fragment layout (f16) in workspace.
// B layout (16-bit, 32x16 K-x-N): lane = {g=lane>>4, n=lane&15}; half h -> K=16*(h>>3)+(h&7)+8*g.
// ---------------------------------------------------------------------------
__global__ void setup_kernel(const float* __restrict__ W_rel, const float* __restrict__ b_rel,
                             const float* __restrict__ Wq,  const float* __restrict__ Wk,
                             const float* __restrict__ Wv,  const float* __restrict__ Wo,
                             const float* __restrict__ W1,  const float* __restrict__ W2,
                             const float* __restrict__ Wp,  const float* __restrict__ bp,
                             _Float16* __restrict__ frags, float* __restrict__ pe,
                             float* __restrict__ pbias) {
  const int t = threadIdx.x;

  // Positional encoding pe[9][22]: 11 (sin,cos) pairs per position.
  for (int idx = t; idx < 9 * 11; idx += 256) {
    const int p = idx / 11, k2 = idx % 11;
    const float div = expf((float)(2 * k2) * (-logf(10000.0f) / 22.0f));
    const float ang = (float)p * div;
    pe[p * 22 + 2 * k2]     = sinf(ang);
    pe[p * 22 + 2 * k2 + 1] = cosf(ang);
  }

  // pbias[h][i][j]: bins are integers -9..9, diff i-j in [-8,8] -> bin = i-j+9.
  for (int idx = t; idx < 2 * 81; idx += 256) {
    const int h = idx / 81, ij = idx % 81, i = ij / 9, j = ij % 9;
    const int bin = i - j + 9;
    float acc = bp[h];
    for (int d = 0; d < 22; ++d)
      acc += (W_rel[bin * 22 + d] + b_rel[d]) * Wp[d * 2 + h];
    pbias[idx] = acc;  // idx == (h*9+i)*9 + j
  }

  // Fragment table:
  //   0..2  Wq  (K22,N44, 3 n-tiles)     3..5 Wk     6..8 Wv
  //   9..12 Wo  (K44,N22, 2 k-steps x 2 n-tiles)
  //  13..20 W1  (K22,N128, 8 n-tiles)
  //  21..28 W2  (K128,N22, 4 k-steps x 2 n-tiles)
  for (int idx = t; idx < FRAG_HALFS; idx += 256) {
    const int fi = idx >> 9, lane = (idx >> 4) & 31, h = idx & 15;
    const float* W; int ldw, kbase, nbase, kmax, nmax;
    if (fi < 9)       { const int mi = fi / 3, nt = fi % 3;
                        W = (mi == 0) ? Wq : (mi == 1) ? Wk : Wv;
                        ldw = 44; kbase = 0; nbase = nt * 16; kmax = 22; nmax = 44; }
    else if (fi < 13) { const int q = fi - 9;
                        W = Wo; ldw = 22; kbase = (q >> 1) * 32; nbase = (q & 1) * 16;
                        kmax = 44; nmax = 22; }
    else if (fi < 21) { const int nt = fi - 13;
                        W = W1; ldw = 128; kbase = 0; nbase = nt * 16;
                        kmax = 22; nmax = 128; }
    else              { const int q = fi - 21;
                        W = W2; ldw = 22; kbase = (q >> 1) * 32; nbase = (q & 1) * 16;
                        kmax = 128; nmax = 22; }
    const int k = kbase + ((h >> 3) << 4) + (h & 7) + ((lane >> 4) << 3);   // K = 16*(h>>3) + (h&7) + 8*g within step
    const int n = nbase + (lane & 15);
    const float v = (k < kmax && n < nmax) ? W[k * ldw + n] : 0.0f;
    frags[idx] = (_Float16)v;
  }
}

// ---------------------------------------------------------------------------
// Main fused kernel: 8192 workgroups x 288 threads (9 waves).
// ---------------------------------------------------------------------------
__global__ __launch_bounds__(288)
void fused_model_kernel(const float* __restrict__ seq,
                        const float* __restrict__ g_att, const float* __restrict__ b_att,
                        const float* __restrict__ b1,    const float* __restrict__ b2,
                        const float* __restrict__ g_ff,  const float* __restrict__ b_ff,
                        const float* __restrict__ W_res, const float* __restrict__ b_res,
                        const float* __restrict__ W_out, const float* __restrict__ b_out,
                        const _Float16* __restrict__ frags,
                        const float* __restrict__ pe, const float* __restrict__ pbias,
                        float* __restrict__ out) {
  extern __shared__ float smem[];
  float* xs = smem + SM_XS;   // [144][24]
  float* qs = smem + SM_QS;   // [144][48]
  float* ks = smem + SM_KS;   // [144][48]
  float* vs = smem + SM_VS;   // [144][48]
  float* hs = smem + SM_HS;   // [144][48]
  float* ff = smem + SM_QS;   // [144][132] aliases q/k/v
  float* y1 = smem + SM_KS;   // [144][24]  aliases k
  float* y2 = smem + SM_HS;   // [144][24]  aliases heads
  float* pr = smem + SM_KS;   // [144]      aliases y1

  const int t = threadIdx.x;
  const int wave = t >> 5, lane = t & 31;
  const int rowbase = wave * 16;            // M-tile owned by this wave
  const int nlo = lane & 15;
  const int mhi = (lane >> 4) << 3;         // +8 rows for high lane group (C/D layout)
  const long base = (long)blockIdx.x * (144L * 22L);
  const v16h* bfr = (const v16h*)frags;

  // ---- stage 0: load x tile (contiguous 144*22 floats) + positional enc ----
  for (int idx = t; idx < 144 * 22; idx += 288) {
    const int r = idx / 22, d = idx - r * 22;
    xs[r * 24 + d] = seq[base + idx] + pe[(r % 9) * 22 + d];
  }
  __syncthreads();

  // ---- stage 1: fused QKV projection (9 WMMAs / wave, A reused) ----
  {
    const v16h a = load_a_frag(xs, 24, rowbase, lane, 0, 22);
#pragma unroll
    for (int fi = 0; fi < 9; ++fi) {
      const v16h b = bfr[fi * 32 + lane];
      v8f c = {};
      c = wmma16(a, b, c);
      const int mi = fi / 3, nt = fi % 3;
      float* dst = (mi == 0) ? qs : (mi == 1) ? ks : vs;
      const int n = nt * 16 + nlo;
      if (n < 44) {
        float* drow = dst + (rowbase + mhi) * 48 + n;
#pragma unroll
        for (int r = 0; r < 8; ++r) drow[r * 48] = c[r];
      }
    }
  }
  __syncthreads();

  // ---- stage 2: attention softmax. 288 thr = 16 batch x 2 head x 9 query ----
  {
    const int bb = t / 18, rem = t % 18;
    const int hh = rem / 9, qi = rem % 9;
    const int rb = bb * 9, co = hh * 22;
    float qv[22];
    const float* qrow = qs + (rb + qi) * 48 + co;
#pragma unroll
    for (int d = 0; d < 22; ++d) qv[d] = qrow[d];
    float sc[9], mx = -1e30f;
#pragma unroll
    for (int j = 0; j < 9; ++j) {
      const float* krow = ks + (rb + j) * 48 + co;
      float acc = 0.0f;
#pragma unroll
      for (int d = 0; d < 22; ++d) acc += qv[d] * krow[d];
      sc[j] = acc * 0.21320071635561044f + pbias[(hh * 9 + qi) * 9 + j];
      mx = fmaxf(mx, sc[j]);
    }
    float se = 0.0f;
#pragma unroll
    for (int j = 0; j < 9; ++j) { sc[j] = __expf(sc[j] - mx); se += sc[j]; }
    const float inv = 1.0f / se;
    float* hrow = hs + (rb + qi) * 48 + co;
#pragma unroll
    for (int d = 0; d < 22; ++d) {
      float acc = 0.0f;
#pragma unroll
      for (int j = 0; j < 9; ++j) acc += sc[j] * vs[(rb + j) * 48 + co + d];
      hrow[d] = acc * inv;
    }
  }
  __syncthreads();

  // ---- stage 3: output projection heads@Wo (K=44 -> 2 k-steps, 2 n-tiles) ----
  {
    const v16h a0 = load_a_frag(hs, 48, rowbase, lane, 0, 44);
    const v16h a1 = load_a_frag(hs, 48, rowbase, lane, 32, 44);
#pragma unroll
    for (int nt = 0; nt < 2; ++nt) {
      v8f c = {};
      c = wmma16(a0, bfr[(9 + nt) * 32 + lane], c);
      c = wmma16(a1, bfr[(11 + nt) * 32 + lane], c);
      const int n = nt * 16 + nlo;
      if (n < 22) {
        float* drow = y1 + (rowbase + mhi) * 24 + n;
#pragma unroll
        for (int r = 0; r < 8; ++r) drow[r * 24] = c[r];
      }
    }
  }
  __syncthreads();

  // ---- stage 4: residual + LayerNorm (attention) -> x1 in place ----
  if (t < 144) {
    float v[22], s = 0.0f;
#pragma unroll
    for (int d = 0; d < 22; ++d) { v[d] = xs[t * 24 + d] + y1[t * 24 + d]; s += v[d]; }
    const float m = s * (1.0f / 22.0f);
    float var = 0.0f;
#pragma unroll
    for (int d = 0; d < 22; ++d) { const float dd = v[d] - m; var += dd * dd; }
    const float rstd = rsqrtf(var * (1.0f / 22.0f) + 1e-5f);
#pragma unroll
    for (int d = 0; d < 22; ++d)
      xs[t * 24 + d] = (v[d] - m) * rstd * g_att[d] + b_att[d];
  }
  __syncthreads();

  // ---- stage 5: FF up-projection + bias + ReLU (8 WMMAs, A reused) ----
  {
    const v16h a = load_a_frag(xs, 24, rowbase, lane, 0, 22);
#pragma unroll
    for (int nt = 0; nt < 8; ++nt) {
      v8f c = {};
      c = wmma16(a, bfr[(13 + nt) * 32 + lane], c);
      const int n = nt * 16 + nlo;
      const float bias = b1[n];
      float* drow = ff + (rowbase + mhi) * 132 + n;
#pragma unroll
      for (int r = 0; r < 8; ++r) drow[r * 132] = fmaxf(c[r] + bias, 0.0f);
    }
  }
  __syncthreads();

  // ---- stage 6: FF down-projection (K=128 -> 4 k-steps, 2 n-tiles) ----
  {
    v16h a[4];
#pragma unroll
    for (int ksi = 0; ksi < 4; ++ksi)
      a[ksi] = load_a_frag(ff, 132, rowbase, lane, ksi * 32, 128);
#pragma unroll
    for (int nt = 0; nt < 2; ++nt) {
      v8f c = {};
#pragma unroll
      for (int ksi = 0; ksi < 4; ++ksi)
        c = wmma16(a[ksi], bfr[(21 + ksi * 2 + nt) * 32 + lane], c);
      const int n = nt * 16 + nlo;
      if (n < 22) {
        float* drow = y2 + (rowbase + mhi) * 24 + n;
#pragma unroll
        for (int r = 0; r < 8; ++r) drow[r * 24] = c[r];
      }
    }
  }
  __syncthreads();

  // ---- stage 7: residual + b2 + LayerNorm (FF) -> x2 in place ----
  if (t < 144) {
    float v[22], s = 0.0f;
#pragma unroll
    for (int d = 0; d < 22; ++d) {
      v[d] = xs[t * 24 + d] + y2[t * 24 + d] + b2[d];
      s += v[d];
    }
    const float m = s * (1.0f / 22.0f);
    float var = 0.0f;
#pragma unroll
    for (int d = 0; d < 22; ++d) { const float dd = v[d] - m; var += dd * dd; }
    const float rstd = rsqrtf(var * (1.0f / 22.0f) + 1e-5f);
#pragma unroll
    for (int d = 0; d < 22; ++d)
      xs[t * 24 + d] = (v[d] - m) * rstd * g_ff[d] + b_ff[d];
  }
  __syncthreads();

  // ---- stage 8: residue head p_res = x2 @ W_res + b_res ----
  if (t < 144) {
    float acc = b_res[0];
#pragma unroll
    for (int d = 0; d < 22; ++d) acc += xs[t * 24 + d] * W_res[d];
    pr[t] = acc;
  }
  __syncthreads();

  // ---- stage 9: output head over sequence dim: [16,9] @ [9,2] + b_out ----
  if (t < 32) {
    const int bb = t >> 1, cc = t & 1;
    float acc = b_out[cc];
#pragma unroll
    for (int l = 0; l < 9; ++l) acc += pr[bb * 9 + l] * W_out[l * 2 + cc];
    out[((long)blockIdx.x * 16 + bb) * 2 + cc] = acc;
  }
}

// ---------------------------------------------------------------------------
extern "C" void kernel_launch(void* const* d_in, const int* in_sizes, int n_in,
                              void* d_out, int out_size, void* d_ws, size_t ws_size,
                              hipStream_t stream) {
  const float* seq   = (const float*)d_in[0];
  const float* W_rel = (const float*)d_in[1];
  const float* b_rel = (const float*)d_in[2];
  const float* Wq    = (const float*)d_in[3];
  const float* Wk    = (const float*)d_in[4];
  const float* Wv    = (const float*)d_in[5];
  const float* Wo    = (const float*)d_in[6];
  const float* g_att = (const float*)d_in[7];
  const float* b_att = (const float*)d_in[8];
  const float* W1    = (const float*)d_in[9];
  const float* b1    = (const float*)d_in[10];
  const float* W2    = (const float*)d_in[11];
  const float* b2    = (const float*)d_in[12];
  const float* Wp    = (const float*)d_in[13];
  const float* bp    = (const float*)d_in[14];
  const float* g_ff  = (const float*)d_in[15];
  const float* b_ff  = (const float*)d_in[16];
  const float* W_res = (const float*)d_in[17];
  const float* b_res = (const float*)d_in[18];
  const float* W_out = (const float*)d_in[19];
  const float* b_out = (const float*)d_in[20];

  _Float16* frags = (_Float16*)d_ws;
  float* pe    = (float*)((char*)d_ws + WS_PE_OFF);
  float* pbias = (float*)((char*)d_ws + WS_PB_OFF);

  setup_kernel<<<1, 256, 0, stream>>>(W_rel, b_rel, Wq, Wk, Wv, Wo, W1, W2,
                                      Wp, bp, frags, pe, pbias);

  const int nblocks = 131072 / 16;             // 16 batch items per workgroup
  const size_t lds_bytes = SM_FLOATS * sizeof(float);  // 124416 B (aliased pool)
  fused_model_kernel<<<nblocks, 288, lds_bytes, stream>>>(
      seq, g_att, b_att, b1, b2, g_ff, b_ff, W_res, b_res, W_out, b_out,
      frags, pe, pbias, (float*)d_out);
}
